// QuantumReIDModel_41111426957521
// MI455X (gfx1250) — matrix operand order repaired and checked by reading, not compiled, not run on hardware
//
#include <hip/hip_runtime.h>
#include <math.h>

// ---------------- problem constants ----------------
#define BATCH   128
#define K_IN    98304
#define HID     128
#define NCIRC   12
#define NLAYERS 12
#define NQ      8
#define NFEAT   (NCIRC * NQ)   // 96
#define NCLS    200
#define KC      192                   // split-K chunks
#define KSTEP   32                    // bf16 WMMA K per step
#define KCHUNK  (K_IN / KC)           // 512
#define STEPS   (KCHUNK / KSTEP)      // 16
#define NMACRO  16                    // 4x4 grid of 32x32 macro-tiles
#define NWAVES  (NMACRO * KC)         // 3072

typedef __attribute__((ext_vector_type(16))) __bf16    v16bf;
typedef __attribute__((ext_vector_type(8)))  float     v8f;
typedef __attribute__((ext_vector_type(8)))  unsigned  v8u;

// ---------------- bf16 hi/lo split helpers ----------------
// hi = truncate-to-bf16 (exact residual), lo = truncate(f - hi).
// Packs 2 bf16 per dword with one v_perm_b32: ~3 VALU per element.
struct BfPair { v16bf hi, lo; };

__device__ __forceinline__ BfPair split16(const float f[16]) {
  v8u h, l;
#pragma unroll
  for (int p = 0; p < 8; ++p) {
    unsigned u0 = __builtin_bit_cast(unsigned, f[2 * p]);
    unsigned u1 = __builtin_bit_cast(unsigned, f[2 * p + 1]);
    // dword = [u1.hi16, u0.hi16]  (bytes 7,6,3,2 of {u1,u0})
    h[p] = __builtin_amdgcn_perm(u1, u0, 0x07060302u);
    float r0 = f[2 * p]     - __builtin_bit_cast(float, u0 & 0xFFFF0000u);
    float r1 = f[2 * p + 1] - __builtin_bit_cast(float, u1 & 0xFFFF0000u);
    l[p] = __builtin_amdgcn_perm(__builtin_bit_cast(unsigned, r1),
                                 __builtin_bit_cast(unsigned, r0), 0x07060302u);
  }
  BfPair o;
  o.hi = __builtin_bit_cast(v16bf, h);
  o.lo = __builtin_bit_cast(v16bf, l);
  return o;
}

// A fragment: 16x32 f32 tile, lane = M; two contiguous 8-float runs (ISA 7.12.2)
__device__ __forceinline__ void loadA(const float* __restrict__ row, int kk,
                                      int khalf, float f[16]) {
#pragma unroll
  for (int i = 0; i < 8; ++i) f[i]     = row[kk + khalf * 8 + i];
#pragma unroll
  for (int i = 0; i < 8; ++i) f[8 + i] = row[kk + 16 + khalf * 8 + i];
}
// B fragment: 32x16 f32 tile, lane = N; two contiguous 8-float runs (same k order as A)
__device__ __forceinline__ void loadB(const float* __restrict__ row, int kk,
                                      int khalf, float f[16]) {
#pragma unroll
  for (int i = 0; i < 8; ++i) f[i]     = row[kk + khalf * 8 + i];
#pragma unroll
  for (int i = 0; i < 8; ++i) f[8 + i] = row[kk + 16 + khalf * 8 + i];
}

#define WMMA_BF16(A, B, C) \
  __builtin_amdgcn_wmma_f32_16x16x32_bf16(false, (A), false, (B), (short)0, (C), false, false)

// ---------------- kernel 0: feats = bias ----------------
__global__ void init_feats_kernel(const float* __restrict__ b_fr,
                                  float* __restrict__ feats) {
  int i = blockIdx.x * blockDim.x + threadIdx.x;   // 16384 total
  feats[i] = b_fr[i & (HID - 1)];
}

// ---------------- kernel 1: split-K bf16x3 WMMA GEMM  feats += x @ W_fr^T ----
// wave computes a 32x32 output macro-tile slice over one k-chunk:
// 2 A-frags x 2 B-frags -> 4 accumulators, 12 WMMAs per k-step.
__global__ void gemm_feats_kernel(const float* __restrict__ x,
                                  const float* __restrict__ Wfr,
                                  float* __restrict__ feats) {
  const int wave  = (blockIdx.x * blockDim.x + threadIdx.x) >> 5;
  const int lane  = threadIdx.x & 31;
  const int macro = wave & (NMACRO - 1);   // 4x4 macro-tiles
  const int kc    = wave >> 4;             // 192 k-chunks
  const int mt    = (macro >> 2) << 5;
  const int nt    = (macro & 3) << 5;
  const int khalf = lane >> 4;
  const int l15   = lane & 15;

  const float* __restrict__ xrow0 = x   + (size_t)(mt + l15) * K_IN;
  const float* __restrict__ xrow1 = x   + (size_t)(mt + 16 + l15) * K_IN;
  const float* __restrict__ wrow0 = Wfr + (size_t)(nt + l15) * K_IN;
  const float* __restrict__ wrow1 = Wfr + (size_t)(nt + 16 + l15) * K_IN;

  v8f acc00, acc01, acc10, acc11;
#pragma unroll
  for (int r = 0; r < 8; ++r) { acc00[r] = 0.f; acc01[r] = 0.f; acc10[r] = 0.f; acc11[r] = 0.f; }

  int kk = kc * KCHUNK;
  for (int s = 0; s < STEPS; ++s, kk += KSTEP) {
    // branch-free clamped prefetch, WGP-near (fills all cache levels)
    int kpf = kk + 4 * KSTEP;
    kpf = (kpf > K_IN - KSTEP) ? (K_IN - KSTEP) : kpf;
    __builtin_prefetch(&xrow0[kpf], 0, 3);
    __builtin_prefetch(&xrow1[kpf], 0, 3);
    __builtin_prefetch(&wrow0[kpf], 0, 3);
    __builtin_prefetch(&wrow1[kpf], 0, 3);

    float fa[16];
    loadA(xrow0, kk, khalf, fa);  BfPair a0 = split16(fa);
    loadA(xrow1, kk, khalf, fa);  BfPair a1 = split16(fa);
    loadB(wrow0, kk, khalf, fa);  BfPair b0 = split16(fa);
    loadB(wrow1, kk, khalf, fa);  BfPair b1 = split16(fa);

    // bf16x3 fp32 emulation per tile: hi*hi + hi*lo + lo*hi
    acc00 = WMMA_BF16(a0.hi, b0.hi, acc00);
    acc00 = WMMA_BF16(a0.hi, b0.lo, acc00);
    acc00 = WMMA_BF16(a0.lo, b0.hi, acc00);

    acc01 = WMMA_BF16(a0.hi, b1.hi, acc01);
    acc01 = WMMA_BF16(a0.hi, b1.lo, acc01);
    acc01 = WMMA_BF16(a0.lo, b1.hi, acc01);

    acc10 = WMMA_BF16(a1.hi, b0.hi, acc10);
    acc10 = WMMA_BF16(a1.hi, b0.lo, acc10);
    acc10 = WMMA_BF16(a1.lo, b0.hi, acc10);

    acc11 = WMMA_BF16(a1.hi, b1.hi, acc11);
    acc11 = WMMA_BF16(a1.hi, b1.lo, acc11);
    acc11 = WMMA_BF16(a1.lo, b1.hi, acc11);
  }

  // C/D layout: VGPR r -> M = khalf*8 + r, N = lane&15
  const int mrow = mt + khalf * 8;
  const int ncol = nt + l15;
#pragma unroll
  for (int r = 0; r < 8; ++r) {
    atomicAdd(&feats[(mrow + r) * HID + ncol],           acc00[r]);
    atomicAdd(&feats[(mrow + r) * HID + ncol + 16],      acc01[r]);
    atomicAdd(&feats[(mrow + 16 + r) * HID + ncol],      acc10[r]);
    atomicAdd(&feats[(mrow + 16 + r) * HID + ncol + 16], acc11[r]);
  }
}

// ---------------- kernel 2: quantum circuit state-vector sim ----------------
// one block (128 threads) per (batch b, circuit c); 256-amp state in LDS
__global__ void quantum_kernel(const float* __restrict__ feats,
                               const float* __restrict__ qw,
                               float* __restrict__ h) {
  __shared__ float sre[256], sim_[256], red[128];
  const int t = threadIdx.x;            // 0..127
  const int b = blockIdx.x;             // 0..127
  const int c = blockIdx.y;             // 0..11

  // amplitude embedding: features [c*10, c*10+10) padded to 256, normalized
  float v0 = (t < 10) ? feats[b * HID + c * 10 + t] : 0.0f;
  red[t] = v0 * v0;
  __syncthreads();
  for (int off = 64; off > 0; off >>= 1) {
    if (t < off) red[t] += red[t + off];
    __syncthreads();
  }
  float rn = rsqrtf(red[0]);
  sre[t] = v0 * rn;  sim_[t] = 0.0f;
  sre[t + 128] = 0.0f; sim_[t + 128] = 0.0f;

  for (int l = 0; l < NLAYERS; ++l) {
    // Rot gates: RZ(omega) RY(theta) RZ(phi) on each wire
    for (int q = 0; q < NQ; ++q) {
      const float* w = qw + (((c * NLAYERS) + l) * NQ + q) * 3;
      float phi = w[0], th = w[1], om = w[2];
      float st, ct;  sincosf(0.5f * th, &st, &ct);
      float sap, cap; sincosf(0.5f * (phi + om), &sap, &cap);
      float sam, cam; sincosf(0.5f * (phi - om), &sam, &cam);
      float u00r =  ct * cap, u00i = -ct * sap;
      float u01r = -st * cam, u01i = -st * sam;
      float u10r =  st * cam, u10i = -st * sam;
      float u11r =  ct * cap, u11i =  ct * sap;

      int mask = 1 << (7 - q);          // wire q = bit (7-q)
      int low  = t & (mask - 1);
      int i0   = ((t ^ low) << 1) | low;
      int i1   = i0 | mask;
      __syncthreads();
      float a0r = sre[i0], a0i = sim_[i0];
      float a1r = sre[i1], a1i = sim_[i1];
      sre[i0]  = u00r * a0r - u00i * a0i + u01r * a1r - u01i * a1i;
      sim_[i0] = u00r * a0i + u00i * a0r + u01r * a1i + u01i * a1r;
      sre[i1]  = u10r * a0r - u10i * a0i + u11r * a1r - u11i * a1i;
      sim_[i1] = u10r * a0i + u10i * a0r + u11r * a1i + u11i * a1r;
    }
    // entangling CNOT ring, range r = l % 7 + 1
    int r = (l % (NQ - 1)) + 1;
    for (int q = 0; q < NQ; ++q) {
      int tgt   = (q + r) & 7;
      int maskT = 1 << (7 - tgt);
      int maskC = 1 << (7 - q);
      int low   = t & (maskT - 1);
      int i0    = ((t ^ low) << 1) | low;
      int i1    = i0 | maskT;
      __syncthreads();
      if (i0 & maskC) {
        float tr = sre[i0], ti = sim_[i0];
        sre[i0] = sre[i1]; sim_[i0] = sim_[i1];
        sre[i1] = tr;      sim_[i1] = ti;
      }
    }
  }
  __syncthreads();

  // <Z_q> = sum_s |psi_s|^2 * (1 - 2*bit_q(s))
  float p0 = sre[t] * sre[t] + sim_[t] * sim_[t];
  float p1 = sre[t + 128] * sre[t + 128] + sim_[t + 128] * sim_[t + 128];
  for (int q = 0; q < NQ; ++q) {
    int b0 = (t >> (7 - q)) & 1;
    int b1 = ((t + 128) >> (7 - q)) & 1;
    float s = p0 * (1.0f - 2.0f * b0) + p1 * (1.0f - 2.0f * b1);
    __syncthreads();
    red[t] = s;
    __syncthreads();
    for (int off = 64; off > 0; off >>= 1) {
      if (t < off) red[t] += red[t + off];
      __syncthreads();
    }
    if (t == 0) h[b * NFEAT + c * NQ + q] = red[0];
  }
}

// ---------------- kernel 3a: BN train-mode stats -> scale/shift -------------
__global__ void bn_stats_kernel(const float* __restrict__ h,
                                const float* __restrict__ gamma,
                                const float* __restrict__ beta,
                                float* __restrict__ scale,
                                float* __restrict__ shift) {
  int j = threadIdx.x;
  if (j >= NFEAT) return;
  float s = 0.0f, sq = 0.0f;
  for (int i = 0; i < BATCH; ++i) {
    float v = h[i * NFEAT + j];
    s += v; sq += v * v;
  }
  float mean = s * (1.0f / BATCH);
  float var  = sq * (1.0f / BATCH) - mean * mean;   // biased, as reference
  float sc   = gamma[j] * rsqrtf(var + 1e-5f);
  scale[j] = sc;
  shift[j] = beta[j] - mean * sc;
}

// ---------------- kernel 3b: out = BN(h) @ W_fc^T + b_fc --------------------
__global__ void fc_kernel(const float* __restrict__ h,
                          const float* __restrict__ scale,
                          const float* __restrict__ shift,
                          const float* __restrict__ Wfc,
                          const float* __restrict__ bfc,
                          float* __restrict__ out) {
  int idx = blockIdx.x * blockDim.x + threadIdx.x;
  if (idx >= BATCH * NCLS) return;
  int i = idx / NCLS, k = idx % NCLS;
  float acc = bfc[k];
#pragma unroll 4
  for (int j = 0; j < NFEAT; ++j)
    acc += (h[i * NFEAT + j] * scale[j] + shift[j]) * Wfc[k * NFEAT + j];
  out[idx] = acc;
}

// ---------------- launch --------------------------------------------------
extern "C" void kernel_launch(void* const* d_in, const int* in_sizes, int n_in,
                              void* d_out, int out_size, void* d_ws, size_t ws_size,
                              hipStream_t stream) {
  const float* x     = (const float*)d_in[0];
  const float* Wfr   = (const float*)d_in[1];
  const float* bfr   = (const float*)d_in[2];
  const float* qw    = (const float*)d_in[3];
  const float* gamma = (const float*)d_in[4];
  const float* beta  = (const float*)d_in[5];
  const float* Wfc   = (const float*)d_in[6];
  const float* bfc   = (const float*)d_in[7];
  float* out = (float*)d_out;

  float* ws    = (float*)d_ws;
  float* feats = ws;                       // 128*128   = 16384 floats
  float* h     = ws + 16384;               // 128*96    = 12288 floats
  float* scale = ws + 16384 + 12288;       // 96
  float* shift = scale + NFEAT;            // 96

  // feats = bias
  init_feats_kernel<<<(BATCH * HID) / 256, 256, 0, stream>>>(bfr, feats);

  // feats += x @ W_fr^T  (bf16x3 WMMA, 32x32 macro-tiles, split-K atomic reduce)
  // 16 macro-tiles * 192 k-chunks = 3072 waves -> 384 blocks of 256 (8 waves)
  gemm_feats_kernel<<<NWAVES * 32 / 256, 256, 0, stream>>>(x, Wfr, feats);

  // 1536 state-vector sims
  dim3 qgrid(BATCH, NCIRC);
  quantum_kernel<<<qgrid, 128, 0, stream>>>(feats, qw, h);

  // BN stats + fused normalize/FC
  bn_stats_kernel<<<1, 128, 0, stream>>>(h, gamma, beta, scale, shift);
  fc_kernel<<<(BATCH * NCLS + 255) / 256, 256, 0, stream>>>(h, scale, shift, Wfc, bfc, out);
}
